// LinearAttention_49143015801141
// MI455X (gfx1250) — matrix operand order repaired and checked by reading, not compiled, not run on hardware
//
#include <hip/hip_runtime.h>
#include <hip/hip_bf16.h>

typedef __attribute__((ext_vector_type(16))) _Float16 v16h;
typedef __attribute__((ext_vector_type(8)))  _Float16 v8h;
typedef __attribute__((ext_vector_type(4)))  _Float16 v4h;
typedef __attribute__((ext_vector_type(8)))  float    v8f;

#define Bb 2
#define Tt 4096
#define Ee 512
#define Hh 8
#define Dd 64
#define NC 64   // number of 64-wide chunks along T

#define WMMA_F16(a,b,c) \
  __builtin_amdgcn_wmma_f32_16x16x32_f16(false,(a),false,(b),(short)0,(c),false,false)

// ---- fragment loaders (f16 16x16x32 layouts per CDNA5 ISA §7.12.2) ----
// A 16x32 (M x K), source row-major with `pitch` halves per row.
// lane L: M = L%16 ; g = L/16 ; halves K = {g*8..g*8+7} U {16+g*8..16+g*8+7}
static __device__ __forceinline__ v16h frag_a(const _Float16* base, int pitch) {
  const int lane = threadIdx.x & 31;
  const int m = lane & 15, g = lane >> 4;
  const _Float16* r = base + m * pitch + g * 8;
  v8h lo = *(const v8h*)(r);
  v8h hi = *(const v8h*)(r + 16);
  return __builtin_shufflevector(lo, hi, 0,1,2,3,4,5,6,7,8,9,10,11,12,13,14,15);
}
// B 32x16 (K x N), source stored N-major: row n holds 32 contiguous K halves.
// lane L: N = L%16 ; g = L/16 ; halves K = {g*8..g*8+7} U {16+g*8..16+g*8+7}
static __device__ __forceinline__ v16h frag_b(const _Float16* base, int pitch) {
  const int lane = threadIdx.x & 31;
  const int n = lane & 15, g = lane >> 4;
  const _Float16* r = base + n * pitch + g * 8;
  v8h lo = *(const v8h*)(r);
  v8h hi = *(const v8h*)(r + 16);
  return __builtin_shufflevector(lo, hi, 0,1,2,3,4,5,6,7,8,9,10,11,12,13,14,15);
}

// ---------------------------------------------------------------- K1: f32->f16
__global__ __launch_bounds__(256) void cvt_f32_f16(const float* __restrict__ s,
                                                   _Float16* __restrict__ d, int n4) {
  int i = blockIdx.x * blockDim.x + threadIdx.x;
  if (i < n4) {
    float4 v = ((const float4*)s)[i];
    v4h o = { (_Float16)v.x, (_Float16)v.y, (_Float16)v.z, (_Float16)v.w };
    ((v4h*)d)[i] = o;
  }
}

// ------------------------------------------------- K2: fused QKV projection
// y = x @ W^T + b ; feature map elu(y)+1 on Q,K. Grid: (M/64, H, 3)
__global__ __launch_bounds__(256)
void qkv_kernel(const _Float16* __restrict__ xh,
                const _Float16* __restrict__ Wqh, const float* __restrict__ bq,
                const _Float16* __restrict__ Wkh, const float* __restrict__ bk,
                const _Float16* __restrict__ Wvh, const float* __restrict__ bv,
                _Float16* __restrict__ Qf, _Float16* __restrict__ Kf,
                _Float16* __restrict__ Ktf, _Float16* __restrict__ Vtf) {
  const int which = blockIdx.z;
  const _Float16* W = (which == 0) ? Wqh : (which == 1) ? Wkh : Wvh;
  const float* bias = (which == 0) ? bq : (which == 1) ? bk : bv;
  const int wave = threadIdx.x >> 5, lane = threadIdx.x & 31;
  const int tr = wave & 3, tc0 = (wave >> 2) * 2;
  const int row0 = blockIdx.x * 64 + tr * 16;      // global m (B*T rows)
  const int h = blockIdx.y;
  const int col0 = h * 64 + tc0 * 16;              // global n (E cols)

  v8f acc0 = {}, acc1 = {};
  for (int k = 0; k < Ee; k += 32) {
    v16h a  = frag_a(xh + (size_t)row0 * Ee + k, Ee);
    v16h b0 = frag_b(W + (size_t)col0 * Ee + k, Ee);
    v16h b1 = frag_b(W + (size_t)(col0 + 16) * Ee + k, Ee);
    acc0 = WMMA_F16(a, b0, acc0);
    acc1 = WMMA_F16(a, b1, acc1);
  }
  const int M0 = (lane >> 4) * 8, N = lane & 15;
  const float bias0 = bias[col0 + N], bias1 = bias[col0 + 16 + N];
#pragma unroll
  for (int r = 0; r < 8; ++r) {
    const int m = row0 + M0 + r;
    const int b = m >> 12, t = m & (Tt - 1);
    const int bh = b * Hh + h;
#pragma unroll
    for (int cci = 0; cci < 2; ++cci) {
      float v = (cci == 0 ? acc0[r] + bias0 : acc1[r] + bias1);
      const int dloc = tc0 * 16 + cci * 16 + N;    // 0..63 within head
      if (which <= 1) v = (v > 0.0f) ? v + 1.0f : __expf(v);  // elu+1
      const _Float16 hv = (_Float16)v;
      if (which == 0) {
        Qf[(size_t)(bh * Tt + t) * Dd + dloc] = hv;
      } else if (which == 1) {
        Kf [(size_t)(bh * Tt + t) * Dd + dloc] = hv;
        Ktf[(size_t)(bh * Dd + dloc) * Tt + t] = hv;
      } else {
        Vtf[(size_t)(bh * Dd + dloc) * Tt + t] = hv;
      }
    }
  }
}

// ---------------------------- K3: per-(b,h) sequential chunk KV-state pass
// Stores exclusive-prefix S_c^T (e,d) as f16 and z_c (f32) per chunk.
// S^T = V^T K computed with WMMA; accumulators persist across chunks.
__global__ __launch_bounds__(128)
void state_kernel(const _Float16* __restrict__ Ktf, const _Float16* __restrict__ Vtf,
                  _Float16* __restrict__ Sst, float* __restrict__ zst) {
  const int bh = blockIdx.x;                 // 0..15
  const int wave = threadIdx.x >> 5, lane = threadIdx.x & 31;
  const int er0 = (wave & 1) * 32;           // e quadrant
  const int dc0 = (wave >> 1) * 32;          // d quadrant
  const int tid = threadIdx.x;
  const int M0 = (lane >> 4) * 8, N = lane & 15;

  v8f acc[2][2] = {};
  float zreg = 0.0f;

  for (int c = 0; c < NC; ++c) {
    // 1) spill exclusive state for chunk c (f16, row e contiguous in d)
    _Float16* Sout = Sst + (size_t)(bh * NC + c) * Dd * Dd;
#pragma unroll
    for (int i = 0; i < 2; ++i)
#pragma unroll
      for (int j = 0; j < 2; ++j)
#pragma unroll
        for (int r = 0; r < 8; ++r) {
          const int e = er0 + i * 16 + M0 + r;
          const int d = dc0 + j * 16 + N;
          Sout[e * Dd + d] = (_Float16)acc[i][j][r];
        }
    // 2) exclusive z, then add this chunk's key-sums (waves 0-1, wave-uniform)
    if (tid < Dd) {
      zst[(size_t)(bh * NC + c) * Dd + tid] = zreg;
      const _Float16* kr = Ktf + (size_t)(bh * Dd + tid) * Tt + c * 64;
      float s = 0.0f;
      for (int t2 = 0; t2 < 64; ++t2) s += (float)kr[t2];
      zreg += s;
    }
    // 3) S^T += V^T(A: Vt rows e) @ K(B: Kt rows d), K-dim = 64 time steps
#pragma unroll
    for (int kk = 0; kk < 64; kk += 32) {
      v16h a0 = frag_a(Vtf + (size_t)(bh * Dd + er0     ) * Tt + c * 64 + kk, Tt);
      v16h a1 = frag_a(Vtf + (size_t)(bh * Dd + er0 + 16) * Tt + c * 64 + kk, Tt);
      v16h b0 = frag_b(Ktf + (size_t)(bh * Dd + dc0     ) * Tt + c * 64 + kk, Tt);
      v16h b1 = frag_b(Ktf + (size_t)(bh * Dd + dc0 + 16) * Tt + c * 64 + kk, Tt);
      acc[0][0] = WMMA_F16(a0, b0, acc[0][0]);
      acc[0][1] = WMMA_F16(a0, b1, acc[0][1]);
      acc[1][0] = WMMA_F16(a1, b0, acc[1][0]);
      acc[1][1] = WMMA_F16(a1, b1, acc[1][1]);
    }
  }
}

// --------------------------------- K4: per-chunk attention (fully parallel)
// O = mask(Q K^T) V + Q S_c ; den = q.z_c + rowsum(mask(P)) + eps
__global__ __launch_bounds__(256)
void attn_kernel(const _Float16* __restrict__ Qf, const _Float16* __restrict__ Kf,
                 const _Float16* __restrict__ Vtf, const _Float16* __restrict__ Sst,
                 const float* __restrict__ zst, _Float16* __restrict__ Of) {
  __shared__ __align__(16) _Float16 Pl[64 * 80];  // padded pitch, 16B-aligned rows
  __shared__ float denl[64];

  const int c = blockIdx.x, h = blockIdx.y, b = blockIdx.z;
  const int bh = b * Hh + h;
  const int wave = threadIdx.x >> 5, lane = threadIdx.x & 31;
  const int tr = wave & 3, tc0 = (wave >> 2) * 2;
  const int M0 = (lane >> 4) * 8, N = lane & 15;

  const _Float16* Qc = Qf + (size_t)(bh * Tt + c * 64) * Dd;   // 64x64, pitch 64
  const _Float16* Kc = Kf + (size_t)(bh * Tt + c * 64) * Dd;
  const _Float16* Sc = Sst + (size_t)(bh * NC + c) * Dd * Dd;  // S^T, pitch 64

  // P = Q @ K^T
  v8f p0 = {}, p1 = {};
#pragma unroll
  for (int kk = 0; kk < Dd; kk += 32) {
    v16h a  = frag_a(Qc + (tr * 16) * Dd + kk, Dd);
    v16h b0 = frag_b(Kc + ((tc0    ) * 16) * Dd + kk, Dd);
    v16h b1 = frag_b(Kc + ((tc0 + 1) * 16) * Dd + kk, Dd);
    p0 = WMMA_F16(a, b0, p0);
    p1 = WMMA_F16(a, b1, p1);
  }
  // causal mask + stage f16 scores through LDS (becomes next A operand)
#pragma unroll
  for (int r = 0; r < 8; ++r) {
    const int t = tr * 16 + M0 + r;
    const int u0 = tc0 * 16 + N, u1 = u0 + 16;
    Pl[t * 80 + u0] = (_Float16)((t >= u0) ? p0[r] : 0.0f);
    Pl[t * 80 + u1] = (_Float16)((t >= u1) ? p1[r] : 0.0f);
  }
  __syncthreads();

  // O = Pmask @ V  +  Q @ S_c
  v8f o0 = {}, o1 = {};
#pragma unroll
  for (int kk = 0; kk < 64; kk += 32) {
    v16h a  = frag_a(Pl + (tr * 16) * 80 + kk, 80);
    v16h b0 = frag_b(Vtf + (size_t)(bh * Dd + tc0 * 16     ) * Tt + c * 64 + kk, Tt);
    v16h b1 = frag_b(Vtf + (size_t)(bh * Dd + tc0 * 16 + 16) * Tt + c * 64 + kk, Tt);
    o0 = WMMA_F16(a, b0, o0);
    o1 = WMMA_F16(a, b1, o1);
  }
#pragma unroll
  for (int kk = 0; kk < Dd; kk += 32) {
    v16h a  = frag_a(Qc + (tr * 16) * Dd + kk, Dd);
    v16h b0 = frag_b(Sc + (tc0 * 16     ) * Dd + kk, Dd);
    v16h b1 = frag_b(Sc + (tc0 * 16 + 16) * Dd + kk, Dd);
    o0 = WMMA_F16(a, b0, o0);
    o1 = WMMA_F16(a, b1, o1);
  }

  // denominator (waves 0-1 only; wave-uniform predicate keeps EXEC clean)
  if (threadIdx.x < 64) {
    const int t = threadIdx.x;
    const float* z = zst + (size_t)(bh * NC + c) * Dd;
    const _Float16* qrow = Qc + t * Dd;
    float s = 1e-6f;
    for (int d = 0; d < Dd; ++d) s += (float)qrow[d] * z[d];
    for (int u = 0; u < 64; ++u) s += (float)Pl[t * 80 + u];
    denl[t] = s;
  }
  __syncthreads();

#pragma unroll
  for (int r = 0; r < 8; ++r) {
    const int t = tr * 16 + M0 + r;
    const float invd = 1.0f / denl[t];
    const int e0 = tc0 * 16 + N;
    const size_t orow = (size_t)(b * Tt + c * 64 + t) * Ee + h * 64;
    Of[orow + e0]      = (_Float16)(o0[r] * invd);
    Of[orow + e0 + 16] = (_Float16)(o1[r] * invd);
  }
}

// ---------------------------------------------- K5: output projection (f32 out)
__global__ __launch_bounds__(256)
void out_proj_kernel(const _Float16* __restrict__ Of, const _Float16* __restrict__ Woh,
                     const float* __restrict__ bo, float* __restrict__ out) {
  const int wave = threadIdx.x >> 5, lane = threadIdx.x & 31;
  const int tr = wave & 3, tc0 = (wave >> 2) * 2;
  const int row0 = blockIdx.x * 64 + tr * 16;
  const int col0 = blockIdx.y * 64 + tc0 * 16;
  v8f acc0 = {}, acc1 = {};
  for (int k = 0; k < Ee; k += 32) {
    v16h a  = frag_a(Of + (size_t)row0 * Ee + k, Ee);
    v16h b0 = frag_b(Woh + (size_t)col0 * Ee + k, Ee);
    v16h b1 = frag_b(Woh + (size_t)(col0 + 16) * Ee + k, Ee);
    acc0 = WMMA_F16(a, b0, acc0);
    acc1 = WMMA_F16(a, b1, acc1);
  }
  const int M0 = (lane >> 4) * 8, N = lane & 15;
  const float b0v = bo[col0 + N], b1v = bo[col0 + 16 + N];
#pragma unroll
  for (int r = 0; r < 8; ++r) {
    const int m = row0 + M0 + r;
    out[(size_t)m * Ee + col0 + N]      = acc0[r] + b0v;
    out[(size_t)m * Ee + col0 + 16 + N] = acc1[r] + b1v;
  }
}

// =============================================================== launcher
extern "C" void kernel_launch(void* const* d_in, const int* in_sizes, int n_in,
                              void* d_out, int out_size, void* d_ws, size_t ws_size,
                              hipStream_t stream) {
  const float* x  = (const float*)d_in[0];
  const float* Wq = (const float*)d_in[1];
  const float* bq = (const float*)d_in[2];
  const float* Wk = (const float*)d_in[3];
  const float* bk = (const float*)d_in[4];
  const float* Wv = (const float*)d_in[5];
  const float* bv = (const float*)d_in[6];
  const float* Wo = (const float*)d_in[7];
  const float* bo = (const float*)d_in[8];
  float* out = (float*)d_out;

  char* ws = (char*)d_ws;
  size_t off = 0;
  auto alloc_h = [&](size_t n) { _Float16* p = (_Float16*)(ws + off); off += n * 2; return p; };

  const size_t NXE = (size_t)Bb * Tt * Ee;          // 4,194,304 elements
  const size_t NW  = (size_t)Ee * Ee;               // 262,144
  _Float16* xh  = alloc_h(NXE);
  _Float16* Wqh = alloc_h(NW);
  _Float16* Wkh = alloc_h(NW);
  _Float16* Wvh = alloc_h(NW);
  _Float16* Woh = alloc_h(NW);
  _Float16* Qf  = alloc_h(NXE);                     // (B,H,T,D)
  _Float16* Kf  = alloc_h(NXE);                     // (B,H,T,D)
  _Float16* Ktf = alloc_h(NXE);                     // (B,H,D,T)
  _Float16* Vtf = alloc_h(NXE);                     // (B,H,D,T)
  _Float16* Of  = alloc_h(NXE);                     // (B,T,E)
  _Float16* Sst = alloc_h((size_t)Bb * Hh * NC * Dd * Dd); // chunk states S^T, f16
  off = (off + 255) & ~(size_t)255;
  float* zst = (float*)(ws + off);                  // (B,H,NC,D) f32
  off += (size_t)Bb * Hh * NC * Dd * 4;

  // K1: conversions
  cvt_f32_f16<<<(int)(NXE / 4 / 256), 256, 0, stream>>>(x, xh, (int)(NXE / 4));
  cvt_f32_f16<<<(int)(NW / 4 / 256), 256, 0, stream>>>(Wq, Wqh, (int)(NW / 4));
  cvt_f32_f16<<<(int)(NW / 4 / 256), 256, 0, stream>>>(Wk, Wkh, (int)(NW / 4));
  cvt_f32_f16<<<(int)(NW / 4 / 256), 256, 0, stream>>>(Wv, Wvh, (int)(NW / 4));
  cvt_f32_f16<<<(int)(NW / 4 / 256), 256, 0, stream>>>(Wo, Woh, (int)(NW / 4));

  // K2: fused QKV projection + feature map + layout shuffles
  qkv_kernel<<<dim3(Bb * Tt / 64, Hh, 3), 256, 0, stream>>>(
      xh, Wqh, bq, Wkh, bk, Wvh, bv, Qf, Kf, Ktf, Vtf);

  // K3: chunk KV states (sequential per (b,h))
  state_kernel<<<dim3(Bb * Hh), 128, 0, stream>>>(Ktf, Vtf, Sst, zst);

  // K4: chunked causal linear attention
  attn_kernel<<<dim3(NC, Hh, Bb), 256, 0, stream>>>(Qf, Kf, Vtf, Sst, zst, Of);

  // K5: output projection
  out_proj_kernel<<<dim3(Bb * Tt / 64, Ee / 64), 256, 0, stream>>>(Of, Woh, bo, out);

  (void)in_sizes; (void)n_in; (void)out_size; (void)ws_size;
}
